// MNISTAdderNet_83683142795458
// MI455X (gfx1250) — compile-verified, matrix operand order checked by reading, not run on hardware
//
#include <hip/hip_runtime.h>
#include <hip/hip_bf16.h>

typedef __attribute__((ext_vector_type(16))) _Float16 v16h;
typedef __attribute__((ext_vector_type(8)))  _Float16 v8h;
typedef __attribute__((ext_vector_type(8)))  float    v8f;

#define HW 4096   // 64*64
#define NB 256    // batch

// ---------------------------------------------------------------- reductions
__device__ __forceinline__ float blk_reduce_256(float v, float* red, int tid) {
  red[tid] = v;
  __syncthreads();
  #pragma unroll
  for (int s = 128; s > 0; s >>= 1) {
    if (tid < s) red[tid] += red[tid + s];
    __syncthreads();
  }
  float r = red[0];
  __syncthreads();
  return r;
}

// ---------------------------------------------------------------- layer 1
// x: [256,1,64,64], w1: [16,1,3,3] -> a1 = -sum|patch - w| : [256,16,64,64]
__global__ __launch_bounds__(256) void adder1_kernel(
    const float* __restrict__ x, const float* __restrict__ w1,
    float* __restrict__ a1) {
  __shared__ float xs[18 * 18];
  __shared__ float wst[9 * 16];  // transposed: wst[k*16+o]
  const int tx = threadIdx.x, ty = threadIdx.y;
  const int tid = ty * 16 + tx;
  const int bx = blockIdx.x * 16, by = blockIdx.y * 16;
  const int b = blockIdx.z;

  if (tid < 144) {
    int k = tid >> 4, o = tid & 15;
    wst[tid] = w1[o * 9 + k];
  }
  for (int idx = tid; idx < 324; idx += 256) {
    int iy = idx / 18, ix = idx % 18;
    int gy = by + iy - 1, gx = bx + ix - 1;
    float v = 0.f;
    if (gy >= 0 && gy < 64 && gx >= 0 && gx < 64) v = x[(size_t)b * HW + gy * 64 + gx];
    xs[idx] = v;
  }
  __syncthreads();

  float acc[16];
  #pragma unroll
  for (int o = 0; o < 16; ++o) acc[o] = 0.f;
  #pragma unroll
  for (int k = 0; k < 9; ++k) {
    const int kh = k / 3, kw = k % 3;
    const float xv = xs[(ty + kh) * 18 + (tx + kw)];
    const float4* wp = (const float4*)&wst[k * 16];
    #pragma unroll
    for (int j = 0; j < 4; ++j) {
      float4 w4 = wp[j];
      acc[4 * j + 0] += fabsf(xv - w4.x);
      acc[4 * j + 1] += fabsf(xv - w4.y);
      acc[4 * j + 2] += fabsf(xv - w4.z);
      acc[4 * j + 3] += fabsf(xv - w4.w);
    }
  }
  const int pix = (by + ty) * 64 + (bx + tx);
  #pragma unroll
  for (int o = 0; o < 16; ++o)
    a1[(size_t)(b * 16 + o) * HW + pix] = -acc[o];
}

// ---------------------------------------------------------------- layer 2
// a1 (pre-BN) -> fused BN1+ReLU on tile load -> a2 = -sum|h - w2| : [256,32,64,64]
__global__ __launch_bounds__(256) void adder2_kernel(
    const float* __restrict__ a1, const float* __restrict__ w2,
    const float* __restrict__ sc1, const float* __restrict__ sh1,
    float* __restrict__ a2) {
  __shared__ float hs[16 * 324];   // 16 channels x 18x18 halo tile (20.25 KB)
  __shared__ float wst[144 * 32];  // transposed weights: wst[(c*9+k)*32 + o] (18 KB)
  const int tx = threadIdx.x, ty = threadIdx.y;
  const int tid = ty * 16 + tx;
  const int bx = blockIdx.x * 16, by = blockIdx.y * 16;
  const int b = blockIdx.z;

  for (int idx = tid; idx < 144 * 32; idx += 256) {
    int f = idx >> 5, o = idx & 31;
    wst[idx] = w2[o * 144 + f];
  }
  for (int idx = tid; idx < 16 * 324; idx += 256) {
    int c = idx / 324, r = idx % 324;
    int iy = r / 18, ix = r % 18;
    int gy = by + iy - 1, gx = bx + ix - 1;
    float v = 0.f;  // zero padding is applied to the POST-BN/ReLU feature map
    if (gy >= 0 && gy < 64 && gx >= 0 && gx < 64) {
      float raw = a1[(size_t)(b * 16 + c) * HW + gy * 64 + gx];
      v = fmaxf(0.f, raw * sc1[c] + sh1[c]);
    }
    hs[idx] = v;
  }
  __syncthreads();

  float acc[32];
  #pragma unroll
  for (int o = 0; o < 32; ++o) acc[o] = 0.f;
  for (int c = 0; c < 16; ++c) {
    #pragma unroll
    for (int k = 0; k < 9; ++k) {
      const int kh = k / 3, kw = k % 3;
      const float xv = hs[c * 324 + (ty + kh) * 18 + (tx + kw)];
      const float4* wp = (const float4*)&wst[(c * 9 + k) * 32];
      #pragma unroll
      for (int j = 0; j < 8; ++j) {
        float4 w4 = wp[j];
        acc[4 * j + 0] += fabsf(xv - w4.x);
        acc[4 * j + 1] += fabsf(xv - w4.y);
        acc[4 * j + 2] += fabsf(xv - w4.z);
        acc[4 * j + 3] += fabsf(xv - w4.w);
      }
    }
  }
  const int pix = (by + ty) * 64 + (bx + tx);
  #pragma unroll
  for (int o = 0; o < 32; ++o)
    a2[(size_t)(b * 32 + o) * HW + pix] = -acc[o];
}

// ---------------------------------------------------------------- BN batch stats
// per-channel sum & sumsq over [B, C, HW], vectorized float4; blockIdx.y = channel
__global__ __launch_bounds__(256) void chan_stats_kernel(
    const float* __restrict__ data, int C, float* __restrict__ stats) {
  __shared__ float red[256];
  const int tid = threadIdx.x;
  const int c = blockIdx.y;
  const int total4 = NB * (HW / 4);  // float4 elements per channel
  const float4* __restrict__ d4 = (const float4*)data;
  float s = 0.f, sq = 0.f;
  for (int e = blockIdx.x * 256 + tid; e < total4; e += gridDim.x * 256) {
    int bb = e >> 10, q = e & 1023;
    float4 v = d4[(size_t)(bb * C + c) * (HW / 4) + q];
    s += v.x + v.y + v.z + v.w;
    sq += v.x * v.x + v.y * v.y + v.z * v.z + v.w * v.w;
  }
  float bs = blk_reduce_256(s, red, tid);
  float bq = blk_reduce_256(sq, red, tid);
  if (tid == 0) {
    atomicAdd(&stats[c], bs);
    atomicAdd(&stats[C + c], bq);
  }
}

__global__ void bn_finalize_kernel(const float* __restrict__ stats,
                                   const float* __restrict__ gamma,
                                   const float* __restrict__ beta, int C,
                                   float invcnt, float* __restrict__ sc,
                                   float* __restrict__ sh) {
  int c = threadIdx.x;
  if (c < C) {
    float mean = stats[c] * invcnt;
    float var = stats[C + c] * invcnt - mean * mean;
    float inv = rsqrtf(var + 1e-5f);
    float s = gamma[c] * inv;
    sc[c] = s;
    sh[c] = beta[c] - mean * s;
  }
}

// ---------------------------------------------------------------- pool (BN2+ReLU fused)
__global__ __launch_bounds__(256) void pool_kernel(
    const float* __restrict__ a2, const float* __restrict__ sc2,
    const float* __restrict__ sh2, _Float16* __restrict__ pooled) {
  __shared__ float red[256];
  const int tid = threadIdx.x;
  const int b = blockIdx.x >> 5, o = blockIdx.x & 31;
  const float s = sc2[o], t = sh2[o];
  const float4* __restrict__ p4 =
      (const float4*)(a2 + (size_t)(b * 32 + o) * HW);
  float sum = 0.f;
  for (int i = tid; i < HW / 4; i += 256) {
    float4 v = p4[i];
    sum += fmaxf(0.f, v.x * s + t) + fmaxf(0.f, v.y * s + t) +
           fmaxf(0.f, v.z * s + t) + fmaxf(0.f, v.w * s + t);
  }
  float bs = blk_reduce_256(sum, red, tid);
  if (tid == 0) pooled[b * 32 + o] = (_Float16)(bs * (1.f / HW));
}

// ---------------------------------------------------------------- FC via WMMA
// pooled [256,32] (f16) x fc_w^T [32,10] + fc_b -> out [256,10] f32.
// One wave per 16-row batch tile; K=32 == one v_wmma_f32_16x16x32_f16.
// 16-bit operand layout (ISA 7.12.2): lane-half g holds, per operand, the two
// contiguous 8-half runs K = g*8..g*8+7 (frag[0..7]) and 16+g*8.. (frag[8..15]).
__global__ __launch_bounds__(32) void fc_wmma_kernel(
    const _Float16* __restrict__ pooled, const float* __restrict__ fcw,
    const float* __restrict__ fcb, float* __restrict__ out) {
  const int lane = threadIdx.x;  // 0..31, full wave -> EXEC all ones
  const int m0 = blockIdx.x * 16;
  const int rc = lane & 15;  // A: row within tile; B: output column (class)
  const int g = lane >> 4;   // lane-half selects K sub-range

  // A fragment: two contiguous 16B runs of the pooled row.
  const v8h* __restrict__ pa = (const v8h*)(pooled + (m0 + rc) * 32);
  v8h alo = pa[g];
  v8h ahi = pa[2 + g];
  v16h a;
  #pragma unroll
  for (int h = 0; h < 8; ++h) { a[h] = alo[h]; a[h + 8] = ahi[h]; }

  // B fragment: fc_w row `rc` (zero for padded columns 10..15), cvt f32->f16.
  v16h b;
  if (rc < 10) {
    const float4* __restrict__ f4 = (const float4*)fcw;
    float4 c0 = f4[rc * 8 + 2 * g + 0];
    float4 c1 = f4[rc * 8 + 2 * g + 1];
    float4 c2 = f4[rc * 8 + 4 + 2 * g + 0];
    float4 c3 = f4[rc * 8 + 4 + 2 * g + 1];
    b[0] = (_Float16)c0.x;  b[1] = (_Float16)c0.y;
    b[2] = (_Float16)c0.z;  b[3] = (_Float16)c0.w;
    b[4] = (_Float16)c1.x;  b[5] = (_Float16)c1.y;
    b[6] = (_Float16)c1.z;  b[7] = (_Float16)c1.w;
    b[8] = (_Float16)c2.x;  b[9] = (_Float16)c2.y;
    b[10] = (_Float16)c2.z; b[11] = (_Float16)c2.w;
    b[12] = (_Float16)c3.x; b[13] = (_Float16)c3.y;
    b[14] = (_Float16)c3.z; b[15] = (_Float16)c3.w;
  } else {
    #pragma unroll
    for (int h = 0; h < 16; ++h) b[h] = (_Float16)0.0f;
  }

  v8f c = {};
  c = __builtin_amdgcn_wmma_f32_16x16x32_f16(false, a, false, b, (short)0, c,
                                             false, false);
  if (rc < 10) {
    const float bias = fcb[rc];
    #pragma unroll
    for (int r = 0; r < 8; ++r) {
      const int m = (g << 3) + r;  // D: VGPR r holds M = r + 8*(lane/16)
      out[(m0 + m) * 10 + rc] = c[r] + bias;
    }
  }
}

// ---------------------------------------------------------------- launch
extern "C" void kernel_launch(void* const* d_in, const int* in_sizes, int n_in,
                              void* d_out, int out_size, void* d_ws,
                              size_t ws_size, hipStream_t stream) {
  (void)in_sizes; (void)n_in; (void)out_size; (void)ws_size;
  const float* x   = (const float*)d_in[0];
  const float* w1  = (const float*)d_in[1];
  const float* g1  = (const float*)d_in[2];
  const float* b1  = (const float*)d_in[3];
  const float* w2  = (const float*)d_in[4];
  const float* g2  = (const float*)d_in[5];
  const float* b2  = (const float*)d_in[6];
  const float* fcw = (const float*)d_in[7];
  const float* fcb = (const float*)d_in[8];
  float* out = (float*)d_out;

  char* ws = (char*)d_ws;
  const size_t A1_BYTES = (size_t)NB * 16 * HW * 4;  //  64 MB
  const size_t A2_BYTES = (size_t)NB * 32 * HW * 4;  // 128 MB
  float* a1 = (float*)ws;
  float* a2 = (float*)(ws + A1_BYTES);
  char* sm = ws + A1_BYTES + A2_BYTES;
  float*     stats1 = (float*)(sm + 0);    // 32 f32 (sum16, sumsq16)
  float*     stats2 = (float*)(sm + 128);  // 64 f32
  float*     sc1    = (float*)(sm + 384);  // 16 f32
  float*     sh1    = (float*)(sm + 448);  // 16 f32
  float*     sc2    = (float*)(sm + 512);  // 32 f32
  float*     sh2    = (float*)(sm + 640);  // 32 f32
  _Float16*  pooled = (_Float16*)(sm + 768);  // 256*32 f16

  // zero the atomic accumulators every launch (harness does not re-poison)
  hipMemsetAsync(sm, 0, 384, stream);

  adder1_kernel<<<dim3(4, 4, NB), dim3(16, 16), 0, stream>>>(x, w1, a1);
  chan_stats_kernel<<<dim3(64, 16), 256, 0, stream>>>(a1, 16, stats1);
  bn_finalize_kernel<<<1, 16, 0, stream>>>(stats1, g1, b1, 16,
                                           1.f / (float)(NB * HW), sc1, sh1);
  adder2_kernel<<<dim3(4, 4, NB), dim3(16, 16), 0, stream>>>(a1, w2, sc1, sh1, a2);
  chan_stats_kernel<<<dim3(64, 32), 256, 0, stream>>>(a2, 32, stats2);
  bn_finalize_kernel<<<1, 32, 0, stream>>>(stats2, g2, b2, 32,
                                           1.f / (float)(NB * HW), sc2, sh2);
  pool_kernel<<<NB * 32, 256, 0, stream>>>(a2, sc2, sh2, pooled);
  fc_wmma_kernel<<<16, 32, 0, stream>>>(pooled, fcw, fcb, out);
}
